// SoftMixture_22694607192161
// MI455X (gfx1250) — matrix operand order repaired and checked by reading, not compiled, not run on hardware
//
#include <hip/hip_runtime.h>
#include <hip/hip_bf16.h>
#include <math.h>

// ---------------------------------------------------------------------------
// SoftMixture (linear -> soft mixture pooling -> 2x Mamba) for MI455X gfx1250.
// GEMMs: v_wmma_f32_16x16x32_f16 with f32 accumulation.
// Tile staging: Tensor Data Mover (tensor_load_to_lds, TENSORcnt), double
// buffered; all GEMM operands pre-converted to f16 (weights transposed to
// [N,K]) so every tile is a plain row-major 2D TDM copy, k-contiguous in LDS.
// ---------------------------------------------------------------------------

typedef __attribute__((ext_vector_type(16))) _Float16 v16h;
typedef __attribute__((ext_vector_type(8)))  _Float16 v8h;
typedef __attribute__((ext_vector_type(8)))  float    v8f;
typedef __attribute__((ext_vector_type(4)))  unsigned int u32x4;
typedef __attribute__((ext_vector_type(8)))  int      i32x8;
typedef __attribute__((ext_vector_type(4)))  int      i32x4;

#define BB      8
#define SS      2048
#define D_IN    1024
#define D_OUT   1024
#define LAT     256
#define N_ST    32
#define D_INNER 2048
#define D_CONV  4
#define DT_RANK 64

__device__ __forceinline__ float silu_f(float v)     { return v / (1.0f + __expf(-v)); }
__device__ __forceinline__ float softplus_f(float v) { return (v > 20.0f) ? v : log1pf(__expf(v)); }

// ---------------------------------------------------------------------------
// TDM: stage one 128-row x 32-halfs tile (row stride = row_stride elements)
// into LDS at lds_off.  LDS row = 64B data + 16B pad  (pad_interval = 16
// DWORDs, pad_amount = 4 DWORDs)  ->  matches  _Float16 [128][40] layout.
// ---------------------------------------------------------------------------
__device__ __forceinline__ void tdm_tile(unsigned int lds_off,
                                         const _Float16* gptr, int row_stride) {
    unsigned long long ga = (unsigned long long)(uintptr_t)gptr;
    u32x4 g0;
    g0[0] = 1u;                                            // count=1 (valid D#)
    g0[1] = lds_off;                                       // lds_addr
    g0[2] = (unsigned int)ga;                              // global_addr lo
    g0[3] = (unsigned int)((ga >> 32) & 0x01FFFFFFu) | (2u << 30);  // addr hi | type=2
    unsigned int td0 = (unsigned int)row_stride;           // tensor_dim0
    unsigned int td1 = 1u << 20;                           // tensor_dim1 (large)
    unsigned long long st0 = (unsigned long long)row_stride; // tensor_dim0_stride
    i32x8 g1;
    g1[0] = (int)((1u << 16) | (1u << 20) | (3u << 22) | (3u << 25)); // 2B elems, pad on
    g1[1] = (int)((td0 & 0xFFFFu) << 16);                  // atomic_addr=0 | td0.lo
    g1[2] = (int)(((td0 >> 16) & 0xFFFFu) | ((td1 & 0xFFFFu) << 16));
    g1[3] = (int)(((td1 >> 16) & 0xFFFFu) | (32u << 16));  // td1.hi | tile_dim0=32
    g1[4] = (int)(128u);                                   // tile_dim1=128, tile_dim2=0
    g1[5] = (int)(st0 & 0xFFFFFFFFu);
    g1[6] = (int)((st0 >> 32) & 0xFFFFu);                  // stride hi | dim1_stride=0
    g1[7] = 0;
    i32x4 z4 = (i32x4){0, 0, 0, 0};
#if __clang_major__ >= 23
    i32x8 z8 = (i32x8){0, 0, 0, 0, 0, 0, 0, 0};
    __builtin_amdgcn_tensor_load_to_lds(g0, g1, z4, z4, z8, 0);
#else
    __builtin_amdgcn_tensor_load_to_lds(g0, g1, z4, z4, 0);
#endif
}

// ---------------------------------------------------------------------------
// WMMA GEMM:  C[b] = act(A[b] @ Bt[b]^T + bias)
//   A : f16 [M,K]  row-major (lda)      Bt : f16 [N,K] row-major (ldb)
// Block 128x128, BK=32, 8 waves (2x4), wave tile 64x32 -> 8 accumulators.
// Tiles staged by TDM, double buffered; fragments via ds_load_b128.
// Outputs: SF32 -> f32 C (RESID accumulates into it), SH -> f16 row-major
// copy, SHT -> f16 per-batch column-major copy [b][n][s] (s = m % SS).
// ---------------------------------------------------------------------------
template <int ACT, int RESID, int HASBIAS, int SF32, int SH, int SHT>
__global__ __launch_bounds__(256) void gemm_wmma(
    const _Float16* __restrict__ A, const _Float16* __restrict__ Bt,
    const float* __restrict__ bias,
    float* __restrict__ Cf, _Float16* __restrict__ Ch, _Float16* __restrict__ Cht,
    int M, int N, int K, int lda, int ldb,
    long sA, long sB, long sC)
{
    constexpr int BK = 32, LDK = 40;
    __shared__ __align__(16) _Float16 As[2][128][LDK];
    __shared__ __align__(16) _Float16 Bs[2][128][LDK];

    const int tid  = threadIdx.x;
    const int lane = tid & 31;
    const int wid  = tid >> 5;
    const int wm   = wid >> 2;           // 0..1
    const int wn   = wid & 3;            // 0..3
    const int l15  = lane & 15;
    const int lhf  = lane >> 4;          // 0/1 (upper half-wave)
    const int bm   = blockIdx.y * 128;
    const int bn   = blockIdx.x * 128;

    A  += (long)blockIdx.z * sA;
    Bt += (long)blockIdx.z * sB;

    v8f acc[4][2];
#pragma unroll
    for (int i = 0; i < 4; ++i)
#pragma unroll
        for (int j = 0; j < 2; ++j)
            acc[i][j] = (v8f){0.f, 0.f, 0.f, 0.f, 0.f, 0.f, 0.f, 0.f};

    const int niter = K / BK;
    if (wid == 0) {
        tdm_tile((unsigned int)(uintptr_t)&As[0][0][0], A  + (long)bm * lda, lda);
        tdm_tile((unsigned int)(uintptr_t)&Bs[0][0][0], Bt + (long)bn * ldb, ldb);
    }
    for (int ki = 0; ki < niter; ++ki) {
        const int cur = ki & 1;
        if (wid == 0) __builtin_amdgcn_s_wait_tensorcnt(0);
        __syncthreads();                       // tiles[ki] visible to all waves
        if (wid == 0 && ki + 1 < niter) {      // prefetch next tiles (other buffer)
            const int k1 = (ki + 1) * BK;
            tdm_tile((unsigned int)(uintptr_t)&As[cur ^ 1][0][0], A  + (long)bm * lda + k1, lda);
            tdm_tile((unsigned int)(uintptr_t)&Bs[cur ^ 1][0][0], Bt + (long)bn * ldb + k1, ldb);
        }
        const _Float16* ab = &As[cur][0][0];
        const _Float16* bb = &Bs[cur][0][0];
        v16h afr[4], bfr[2];
#pragma unroll
        for (int mt = 0; mt < 4; ++mt) {       // A frag: lanes 0-15 K0-7/16-23, +8 upper half
            int row = wm * 64 + mt * 16 + l15;
            v8h lo = *(const v8h*)(ab + row * LDK + lhf * 8);
            v8h hi = *(const v8h*)(ab + row * LDK + 16 + lhf * 8);
#pragma unroll
            for (int i = 0; i < 8; ++i) { afr[mt][i] = lo[i]; afr[mt][i + 8] = hi[i]; }
        }
#pragma unroll
        for (int nt = 0; nt < 2; ++nt) {       // B frag: lane=n, K0-7/16-23, +8 upper half
            int col = wn * 32 + nt * 16 + l15;
            v8h lo = *(const v8h*)(bb + col * LDK + lhf * 8);
            v8h hi = *(const v8h*)(bb + col * LDK + 16 + lhf * 8);
#pragma unroll
            for (int i = 0; i < 8; ++i) { bfr[nt][i] = lo[i]; bfr[nt][i + 8] = hi[i]; }
        }
#pragma unroll
        for (int mt = 0; mt < 4; ++mt)
#pragma unroll
            for (int nt = 0; nt < 2; ++nt)
                acc[mt][nt] = __builtin_amdgcn_wmma_f32_16x16x32_f16(
                    false, afr[mt], false, bfr[nt], (short)0, acc[mt][nt], false, false);
        __syncthreads();                       // done with buffer cur before reuse
    }

    // epilogue: C/D layout (VGPR r: M = r + 8*lhf, N = l15)
    float* Cfb = nullptr;
    if (SF32) Cfb = Cf + (long)blockIdx.z * sC;
#pragma unroll
    for (int mt = 0; mt < 4; ++mt)
#pragma unroll
        for (int nt = 0; nt < 2; ++nt)
#pragma unroll
            for (int r = 0; r < 8; ++r) {
                int m = bm + wm * 64 + mt * 16 + r + lhf * 8;
                int n = bn + wn * 32 + nt * 16 + l15;
                float v = acc[mt][nt][r];
                if (HASBIAS) v += bias[n];
                if (ACT == 1) v = silu_f(v);
                if (ACT == 2) v = softplus_f(v);
                long idx = (long)m * N + n;
                if (RESID) v += Cfb[idx];
                if (SF32)  Cfb[idx] = v;
                if (SH)    Ch[idx] = (_Float16)v;
                if (SHT)   Cht[((long)(m >> 11) * N + n) * SS + (m & (SS - 1))] = (_Float16)v;
            }
}

// ---------------------------------------------------------------------------
// f32 -> f16 row-major copy
__global__ __launch_bounds__(256) void f32_to_f16_k(const float* __restrict__ s,
                                                    _Float16* __restrict__ d) {
    long i = (long)blockIdx.x * 256 + threadIdx.x;
    d[i] = (_Float16)s[i];
}

// f32 [K,N] -> f16 [N,K] tiled transpose-convert
__global__ void transpose_h_k(const float* __restrict__ src, _Float16* __restrict__ dst,
                              int K, int N) {
    __shared__ float t[32][33];
    const int k0 = blockIdx.y * 32, n0 = blockIdx.x * 32;
    const int tx = threadIdx.x, ty = threadIdx.y;   // (32, 8)
#pragma unroll
    for (int i = 0; i < 32; i += 8)
        t[ty + i][tx] = src[(long)(k0 + ty + i) * N + (n0 + tx)];
    __syncthreads();
#pragma unroll
    for (int i = 0; i < 32; i += 8)
        dst[(long)(n0 + ty + i) * K + (k0 + tx)] = (_Float16)t[tx][ty + i];
}

// Row softmax stats over last dim (LAT) of w[B*S, LAT]
__global__ __launch_bounds__(256) void rowstats_k(const float* __restrict__ w,
                                                  float* __restrict__ rmax,
                                                  float* __restrict__ rsum) {
    __shared__ float red[256];
    const int row = blockIdx.x, t = threadIdx.x;
    float v = w[(long)row * LAT + t];
    red[t] = v; __syncthreads();
    for (int s = 128; s > 0; s >>= 1) { if (t < s) red[t] = fmaxf(red[t], red[t + s]); __syncthreads(); }
    float m = red[0]; __syncthreads();
    red[t] = __expf(v - m); __syncthreads();
    for (int s = 128; s > 0; s >>= 1) { if (t < s) red[t] += red[t + s]; __syncthreads(); }
    if (t == 0) { rmax[row] = m; rsum[row] = red[0]; }
}

// Column softmax stats over s per (b, l)
__global__ __launch_bounds__(256) void colstats_k(const float* __restrict__ w,
                                                  float* __restrict__ cmax,
                                                  float* __restrict__ csum) {
    __shared__ float redm[4][64];
    __shared__ float reds[4][64];
    const int b  = blockIdx.x;
    const int li = threadIdx.x & 63;
    const int sg = threadIdx.x >> 6;
    const int l  = blockIdx.y * 64 + li;
    const float* base = w + (long)b * SS * LAT + l;
    float m = -3.4e38f;
    for (int s = sg; s < SS; s += 4) m = fmaxf(m, base[(long)s * LAT]);
    redm[sg][li] = m; __syncthreads();
    float mall = fmaxf(fmaxf(redm[0][li], redm[1][li]), fmaxf(redm[2][li], redm[3][li]));
    float sum = 0.f;
    for (int s = sg; s < SS; s += 4) sum += __expf(base[(long)s * LAT] - mall);
    reds[sg][li] = sum; __syncthreads();
    if (sg == 0) {
        cmax[b * LAT + l] = mall;
        csum[b * LAT + l] = reds[0][li] + reds[1][li] + reds[2][li] + reds[3][li];
    }
}

// weights = 0.5*(softmax_row(w)*LAT/S + softmax_col(w)); store f16 TRANSPOSED
// as [b][l][s] so the einsum GEMM reads it as a plain row-major A.
__global__ __launch_bounds__(256) void weights_k(const float* __restrict__ w,
                                                 const float* __restrict__ rmax,
                                                 const float* __restrict__ rsum,
                                                 const float* __restrict__ cmax,
                                                 const float* __restrict__ csum,
                                                 _Float16* __restrict__ wtT) {
    long gid = (long)blockIdx.x * 256 + threadIdx.x;
    int row = (int)(gid >> 8);       // b*S + s
    int l   = (int)(gid & 255);
    int b   = row >> 11;             // S = 2048
    int s   = row & (SS - 1);
    float v  = w[gid];
    float w1 = __expf(v - rmax[row]) / rsum[row] * ((float)LAT / (float)SS);
    float w2 = __expf(v - cmax[b * LAT + l]) / csum[b * LAT + l];
    wtT[((long)(b * LAT + l)) * SS + s] = (_Float16)(0.5f * (w1 + w2));
}

// RMS-norm rows of [rows,1024] -> f16
__global__ __launch_bounds__(256) void rmsnorm_k(const float* __restrict__ x,
                                                 const float* __restrict__ gamma,
                                                 _Float16* __restrict__ y) {
    __shared__ float red[256];
    const long row = blockIdx.x;
    const int  t   = threadIdx.x;
    const float* xr = x + row * D_OUT;
    float v[4], s = 0.f;
#pragma unroll
    for (int i = 0; i < 4; ++i) { v[i] = xr[t + 256 * i]; s += v[i] * v[i]; }
    red[t] = s; __syncthreads();
    for (int st = 128; st > 0; st >>= 1) { if (t < st) red[t] += red[t + st]; __syncthreads(); }
    float scale = 32.0f / fmaxf(sqrtf(red[0]), 1e-12f);   // sqrt(1024)=32
#pragma unroll
    for (int i = 0; i < 4; ++i)
        y[row * D_OUT + t + 256 * i] = (_Float16)(v[i] * scale * gamma[t + 256 * i]);
}

// Depthwise causal conv1d (width 4) + bias + SiLU; dual-store f32 + f16
__global__ __launch_bounds__(256) void conv_silu_k(const float* __restrict__ xz,
                                                   const float* __restrict__ cw,
                                                   const float* __restrict__ cb,
                                                   float* __restrict__ xh,
                                                   _Float16* __restrict__ xh16) {
    const int g = blockIdx.x * 256 + threadIdx.x;   // B*D_INNER
    const int b = g >> 11, d = g & (D_INNER - 1);
    const float w0 = cw[d * 4 + 0], w1 = cw[d * 4 + 1], w2 = cw[d * 4 + 2], w3 = cw[d * 4 + 3];
    const float bb = cb[d];
    const float* src = xz + (long)b * LAT * (2 * D_INNER) + d;
    float x0 = 0.f, x1 = 0.f, x2 = 0.f;
    for (int t = 0; t < LAT; ++t) {
        float x3 = src[(long)t * (2 * D_INNER)];
        float y  = bb + w0 * x0 + w1 * x1 + w2 * x2 + w3 * x3;
        float sv = silu_f(y);
        long  o  = ((long)b * LAT + t) * D_INNER + d;
        xh[o]   = sv;
        xh16[o] = (_Float16)sv;
        x0 = x1; x1 = x2; x2 = x3;
    }
}

// Selective scan: wave32 per (b,d) channel, lane = state n; y stored f16.
__global__ __launch_bounds__(256) void scan_k(const float* __restrict__ dt,
                                              const float* __restrict__ xh,
                                              const float* __restrict__ xdbc,
                                              const float* __restrict__ xz,
                                              const float* __restrict__ A_log,
                                              const float* __restrict__ Dp,
                                              _Float16* __restrict__ y16) {
    const int gw   = blockIdx.x * 8 + (threadIdx.x >> 5);
    const int lane = threadIdx.x & 31;
    const int b = gw >> 11, d = gw & (D_INNER - 1);
    const float An = -__expf(A_log[d * N_ST + lane]);
    const float Dd = Dp[d];
    float h = 0.f;
    for (int t = 0; t < LAT; ++t) {
        const long base = (long)(b * LAT + t);
        float dtv = dt[base * D_INNER + d];
        float xv  = xh[base * D_INNER + d];
        float Bv  = xdbc[base * 128 + DT_RANK + lane];
        float Cv  = xdbc[base * 128 + DT_RANK + N_ST + lane];
        h = __expf(dtv * An) * h + dtv * Bv * xv;
        float p = h * Cv;
#pragma unroll
        for (int m = 16; m > 0; m >>= 1) p += __shfl_xor(p, m, 32);
        if (lane == 0) {
            float zv = xz[base * (2 * D_INNER) + D_INNER + d];
            y16[base * D_INNER + d] = (_Float16)((p + Dd * xv) * silu_f(zv));
        }
    }
}

// ---------------------------------------------------------------------------
extern "C" void kernel_launch(void* const* d_in, const int* in_sizes, int n_in,
                              void* d_out, int out_size, void* d_ws, size_t ws_size,
                              hipStream_t stream) {
    const float* x       = (const float*)d_in[0];
    const float* W_in    = (const float*)d_in[1];
    const float* b_in    = (const float*)d_in[2];
    const float* Wa1     = (const float*)d_in[3];
    const float* ba1     = (const float*)d_in[4];
    const float* Wa2     = (const float*)d_in[5];
    const float* ba2     = (const float*)d_in[6];
    const float* gamma   = (const float*)d_in[7];
    const float* in_w    = (const float*)d_in[8];
    const float* conv_w  = (const float*)d_in[9];
    const float* conv_b  = (const float*)d_in[10];
    const float* xproj_w = (const float*)d_in[11];
    const float* dtproj_w= (const float*)d_in[12];
    const float* dtproj_b= (const float*)d_in[13];
    const float* A_log   = (const float*)d_in[14];
    const float* Dp      = (const float*)d_in[15];
    const float* out_w   = (const float*)d_in[16];
    float* out = (float*)d_out;                          // [B, LAT, D_OUT] f32

    // ---- workspace: persistent f16 transposed weights, then activations ----
    _Float16* WinT = (_Float16*)d_ws;                    // [1024][1024]
    _Float16* Wa1T = WinT + 1024 * 1024;                 // [256][1024]
    _Float16* Wa2T = Wa1T + 256 * 1024;                  // [256][256]
    _Float16* inwT = Wa2T + 256 * 256;                   // 2x [4096][1024]
    _Float16* xpT  = inwT + 2 * 4096 * 1024;             // 2x [128][2048]
    _Float16* dtpT = xpT  + 2 * 128 * 2048;              // 2x [2048][64]
    _Float16* owT  = dtpT + 2 * 2048 * 64;               // 2x [1024][2048]
    char* actb = (char*)(owT + 2 * 1024 * 2048);
    actb = (char*)(((uintptr_t)actb + 255) & ~(uintptr_t)255);

    // phase 1 (dead after einsum GEMM)
    _Float16* out0h  = (_Float16*)actb;                          // [BS][1024]
    _Float16* out0ht = out0h  + (long)BB * SS * D_OUT;           // [B][1024][S]
    _Float16* h1h    = out0ht + (long)BB * SS * D_OUT;           // [BS][256]
    float*    wbuf   = (float*)(h1h + (long)BB * SS * LAT);      // [BS][256] f32
    _Float16* wtsT   = (_Float16*)(wbuf + (long)BB * SS * LAT);  // [B][256][S]
    float*    rmax   = (float*)(wtsT + (long)BB * SS * LAT);
    float*    rsum   = rmax + BB * SS;
    float*    cmax   = rsum + BB * SS;
    float*    csum   = cmax + BB * LAT;
    _Float16* x16    = (_Float16*)(csum + BB * LAT);             // [BS][1024]

    // phase 2 (reuses phase-1 activation region)
    _Float16* u16    = (_Float16*)actb;                          // [2048][1024]
    float*    xz     = (float*)(u16 + (long)BB * LAT * D_OUT);   // [2048][4096]
    float*    xh     = xz + (long)BB * LAT * 2 * D_INNER;        // [2048][2048]
    _Float16* xh16   = (_Float16*)(xh + (long)BB * LAT * D_INNER);
    float*    xdbc   = (float*)(xh16 + (long)BB * LAT * D_INNER);// [2048][128]
    _Float16* xdbc16 = (_Float16*)(xdbc + (long)BB * LAT * 128);
    float*    dtb    = (float*)(xdbc16 + (long)BB * LAT * 128);  // [2048][2048]
    _Float16* y16    = (_Float16*)(dtb + (long)BB * LAT * D_INNER);

    const dim3 blk(256), tblk(32, 8);
    const int MS = BB * SS;       // 16384
    const int ML = BB * LAT;      // 2048

    // ---- one-time operand conversion ----
    f32_to_f16_k<<<dim3((MS * D_IN) / 256), blk, 0, stream>>>(x, x16);
    transpose_h_k<<<dim3(D_OUT / 32, D_IN / 32), tblk, 0, stream>>>(W_in, WinT, D_IN, D_OUT);
    transpose_h_k<<<dim3(LAT / 32, D_OUT / 32), tblk, 0, stream>>>(Wa1, Wa1T, D_OUT, LAT);
    transpose_h_k<<<dim3(LAT / 32, LAT / 32), tblk, 0, stream>>>(Wa2, Wa2T, LAT, LAT);
    for (int i = 0; i < 2; ++i) {
        transpose_h_k<<<dim3((2 * D_INNER) / 32, D_OUT / 32), tblk, 0, stream>>>(
            in_w + (long)i * D_OUT * 2 * D_INNER, inwT + (long)i * 2 * D_INNER * D_OUT, D_OUT, 2 * D_INNER);
        transpose_h_k<<<dim3(128 / 32, D_INNER / 32), tblk, 0, stream>>>(
            xproj_w + (long)i * D_INNER * 128, xpT + (long)i * 128 * D_INNER, D_INNER, 128);
        transpose_h_k<<<dim3(D_INNER / 32, DT_RANK / 32), tblk, 0, stream>>>(
            dtproj_w + (long)i * DT_RANK * D_INNER, dtpT + (long)i * D_INNER * DT_RANK, DT_RANK, D_INNER);
        transpose_h_k<<<dim3(D_OUT / 32, D_INNER / 32), tblk, 0, stream>>>(
            out_w + (long)i * D_INNER * D_OUT, owT + (long)i * D_OUT * D_INNER, D_INNER, D_OUT);
    }

    // G1: out0 = x @ W_in + b_in  -> f16 row copy + f16 col-major copy
    gemm_wmma<0, 0, 1, 0, 1, 1><<<dim3(D_OUT / 128, MS / 128, 1), blk, 0, stream>>>(
        x16, WinT, b_in, nullptr, out0h, out0ht, MS, D_OUT, D_IN, D_IN, D_IN, 0, 0, 0);
    // G2: h1 = silu(out0 @ Wa1 + ba1) -> f16
    gemm_wmma<1, 0, 1, 0, 1, 0><<<dim3(LAT / 128, MS / 128, 1), blk, 0, stream>>>(
        out0h, Wa1T, ba1, nullptr, h1h, nullptr, MS, LAT, D_OUT, D_OUT, D_OUT, 0, 0, 0);
    // G3: w = h1 @ Wa2 + ba2 -> f32
    gemm_wmma<0, 0, 1, 1, 0, 0><<<dim3(LAT / 128, MS / 128, 1), blk, 0, stream>>>(
        h1h, Wa2T, ba2, wbuf, nullptr, nullptr, MS, LAT, LAT, LAT, LAT, 0, 0, 0);
    // softmax stats + blended weights (transposed f16)
    rowstats_k<<<dim3(MS), blk, 0, stream>>>(wbuf, rmax, rsum);
    colstats_k<<<dim3(BB, LAT / 64), blk, 0, stream>>>(wbuf, cmax, csum);
    weights_k<<<dim3((MS * LAT) / 256), blk, 0, stream>>>(wbuf, rmax, rsum, cmax, csum, wtsT);
    // G4 (batched): out[b] = wtsT[b] @ out0ht[b]^T   [256,1024,2048] x8
    gemm_wmma<0, 0, 0, 1, 0, 0><<<dim3(D_OUT / 128, LAT / 128, BB), blk, 0, stream>>>(
        wtsT, out0ht, nullptr, out, nullptr, nullptr, LAT, D_OUT, SS, SS, SS,
        (long)LAT * SS, (long)D_OUT * SS, (long)LAT * D_OUT);

    // ---- 2 Mamba blocks with residual ----
    for (int i = 0; i < 2; ++i) {
        rmsnorm_k<<<dim3(ML), blk, 0, stream>>>(out, gamma + (long)i * D_OUT, u16);
        // in_proj: xz = u @ in_w[i]
        gemm_wmma<0, 0, 0, 1, 0, 0><<<dim3((2 * D_INNER) / 128, ML / 128, 1), blk, 0, stream>>>(
            u16, inwT + (long)i * 2 * D_INNER * D_OUT, nullptr, xz, nullptr, nullptr,
            ML, 2 * D_INNER, D_OUT, D_OUT, D_OUT, 0, 0, 0);
        // depthwise causal conv + silu (f32 + f16)
        conv_silu_k<<<dim3((BB * D_INNER) / 256), blk, 0, stream>>>(
            xz, conv_w + (long)i * D_INNER * D_CONV, conv_b + (long)i * D_INNER, xh, xh16);
        // x_proj: xdbc = xh @ xw[i]  (f32 for scan, f16 for dt GEMM)
        gemm_wmma<0, 0, 0, 1, 1, 0><<<dim3(128 / 128, ML / 128, 1), blk, 0, stream>>>(
            xh16, xpT + (long)i * 128 * D_INNER, nullptr, xdbc, xdbc16, nullptr,
            ML, 128, D_INNER, D_INNER, D_INNER, 0, 0, 0);
        // dt = softplus(xdbc[:, :64] @ dtw[i] + dtb[i])   (lda=128, K=64)
        gemm_wmma<2, 0, 1, 1, 0, 0><<<dim3(D_INNER / 128, ML / 128, 1), blk, 0, stream>>>(
            xdbc16, dtpT + (long)i * D_INNER * DT_RANK, dtproj_b + (long)i * D_INNER,
            dtb, nullptr, nullptr, ML, D_INNER, DT_RANK, 128, DT_RANK, 0, 0, 0);
        // selective scan + D skip + silu(z) gate -> f16
        scan_k<<<dim3((BB * D_INNER) / 8), blk, 0, stream>>>(
            dtb, xh, xdbc, xz, A_log + (long)i * D_INNER * N_ST, Dp + (long)i * D_INNER, y16);
        // out += y @ out_w[i]  (residual)
        gemm_wmma<0, 1, 0, 1, 0, 0><<<dim3(D_OUT / 128, ML / 128, 1), blk, 0, stream>>>(
            y16, owT + (long)i * D_OUT * D_INNER, nullptr, out, nullptr, nullptr,
            ML, D_OUT, D_INNER, D_INNER, D_INNER, 0, 0, 0);
    }
    (void)in_sizes; (void)n_in; (void)out_size; (void)ws_size;
}
